// NEOVisionEmbeddings_9294309229026
// MI455X (gfx1250) — compile-verified, matrix operand order checked by reading, not run on hardware
//
#include <hip/hip_runtime.h>

typedef __attribute__((ext_vector_type(16))) __bf16 v16bf;
typedef __attribute__((ext_vector_type(8)))  __bf16 v8bf;
typedef __attribute__((ext_vector_type(2)))  __bf16 v2bf;
typedef __attribute__((ext_vector_type(16))) float  v16f;
typedef __attribute__((ext_vector_type(8)))  float  v8f;
typedef __attribute__((ext_vector_type(2)))  float  v2f;

#define NTOK 32768
#define KPIX 588
#define KPAD 608   // 19 chunks of 32
#define EMB  1024
#define NLLM 2048
#define MTOK 8192
#define KMRG 4096

#if defined(__AMDGCN__) && __has_builtin(__builtin_amdgcn_global_load_async_to_lds_b128) && __has_builtin(__builtin_amdgcn_s_wait_asynccnt)
#define USE_ASYNC_LDS 1
typedef int v4i_gcc __attribute__((__vector_size__(16)));
typedef __attribute__((address_space(1))) v4i_gcc* g_v4i;   // global int4*
typedef __attribute__((address_space(3))) v4i_gcc* l_v4i;   // LDS int4*
#else
#define USE_ASYNC_LDS 0
#endif

// float -> bf16 (RNE), native convert.
__device__ __forceinline__ unsigned short f2bf(float f) {
  union { __bf16 h; unsigned short u; } c;
  c.h = (__bf16)f;
  return c.u;
}
// Pack two floats -> one dword of bf16 via a single v_cvt_pk_bf16_f32.
__device__ __forceinline__ unsigned int pack2(float lo, float hi) {
  v2f f; f.x = lo; f.y = hi;
  union { v2bf v; unsigned int u; } c;
  c.v = __builtin_convertvector(f, v2bf);
  return c.u;
}

__device__ __forceinline__ float bf_lo(unsigned int u) {
  union { float f; unsigned int u; } v; v.u = u << 16; return v.f;
}
__device__ __forceinline__ float bf_hi(unsigned int u) {
  union { float f; unsigned int u; } v; v.u = u & 0xffff0000u; return v.f;
}

union Frag  { v16bf v; uint4 q[2]; };
union F16U  { v16f  v; float4 q[4]; };
union B16U  { v16bf v; uint4 q[2]; };
union B8U   { v8bf  v; uint4 q;    };

// ---------------------------------------------------------------------------
// Kernel 0: dense_w [o][e][2][2] f32  ->  bf16 [o][pq*1024 + e]
// ---------------------------------------------------------------------------
__global__ __launch_bounds__(256) void k_convert_dw(
    const float* __restrict__ dw, unsigned short* __restrict__ dwb)
{
  const int t  = blockIdx.x * 256 + threadIdx.x;  // NLLM*4*128 threads
  const int o  = t >> 9;
  const int pq = (t >> 7) & 3;
  const int e0 = (t & 127) * 8;
  const long base = (long)o * 4096;
  v8f f;
  #pragma unroll
  for (int i = 0; i < 8; ++i)
    f[i] = dw[base + (long)(e0 + i) * 4 + pq];
  B8U b; b.v = __builtin_convertvector(f, v8bf);
  *(uint4*)(dwb + base + pq * 1024 + e0) = b.q;
}

// ---------------------------------------------------------------------------
// Kernel 1: patch-embed GEMM  [32768 x 588] * [588 x 1024] + bias + GELU.
// ---------------------------------------------------------------------------
__global__ __launch_bounds__(256) void k_patch_embed(
    const float* __restrict__ pix,      // [NTOK][KPIX]
    const float* __restrict__ pw,       // [EMB][KPIX]
    const float* __restrict__ pb,       // [EMB]
    unsigned short* __restrict__ pe)    // [NTOK][EMB] bf16
{
  __shared__ __align__(16) unsigned short As[2][128][32];
  __shared__ __align__(16) unsigned short Bs[2][128][32];

  const int tid  = threadIdx.x;
  const int lane = tid & 31;
  const int wid  = tid >> 5;
  const int m0   = blockIdx.x * 128;
  const int n0   = blockIdx.y * 128;
  const int wm   = (wid & 3) * 32;
  const int wn   = (wid >> 2) * 64;
  const int lrow = tid >> 1;
  const int lk   = (tid & 1) * 16;

  const float* arow = pix + (long)(m0 + lrow) * KPIX;
  const float* brow = pw  + (long)(n0 + lrow) * KPIX;

  v8f acc[2][4];
  #pragma unroll
  for (int mi = 0; mi < 2; ++mi)
    #pragma unroll
    for (int ni = 0; ni < 4; ++ni)
      acc[mi][ni] = (v8f){0.f,0.f,0.f,0.f,0.f,0.f,0.f,0.f};

  const int fr = lane & 15;
  const int kh = (lane >> 4) * 8;

  auto loadseg = [&](const float* row, int k0, F16U& r) {
    const int s = k0 + lk;
    if (k0 + 32 <= KPIX) {               // wave-uniform fast path
      const float4* p = (const float4*)(row + s);
      r.q[0] = p[0]; r.q[1] = p[1]; r.q[2] = p[2]; r.q[3] = p[3];
    } else {                             // only the final chunk
      #pragma unroll
      for (int i = 0; i < 16; ++i) r.v[i] = (s + i < KPIX) ? row[s + i] : 0.0f;
    }
  };
  auto store2 = [&](unsigned short S[][32], const F16U& r) {
    B16U b;
    b.v = __builtin_convertvector(r.v, v16bf);   // 8x v_cvt_pk_bf16_f32
    *(uint4*)&S[lrow][lk]     = b.q[0];
    *(uint4*)&S[lrow][lk + 8] = b.q[1];
  };

  F16U ra, rb;
  loadseg(arow, 0, ra);
  loadseg(brow, 0, rb);

  const int NC = KPAD / 32;              // 19
  for (int kc = 0; kc < NC; ++kc) {
    const int buf = kc & 1;
    store2(As[buf], ra);
    store2(Bs[buf], rb);
    __syncthreads();
    if (kc + 1 < NC) {                   // prefetch next chunk into registers
      loadseg(arow, (kc + 1) * 32, ra);
      loadseg(brow, (kc + 1) * 32, rb);
    }
    Frag af[2], bfr[4];
    #pragma unroll
    for (int mi = 0; mi < 2; ++mi) {
      const unsigned short* s = &As[buf][wm + mi * 16 + fr][0];
      af[mi].q[0] = *(const uint4*)(s + kh);
      af[mi].q[1] = *(const uint4*)(s + 16 + kh);
    }
    #pragma unroll
    for (int ni = 0; ni < 4; ++ni) {
      const unsigned short* s = &Bs[buf][wn + ni * 16 + fr][0];
      bfr[ni].q[0] = *(const uint4*)(s + kh);
      bfr[ni].q[1] = *(const uint4*)(s + 16 + kh);
    }
    #pragma unroll
    for (int mi = 0; mi < 2; ++mi)
      #pragma unroll
      for (int ni = 0; ni < 4; ++ni)
        acc[mi][ni] = __builtin_amdgcn_wmma_f32_16x16x32_bf16(
            false, af[mi].v, false, bfr[ni].v, (short)0, acc[mi][ni],
            false, false);
  }

  const int rbase = (lane >> 4) * 8;
  #pragma unroll
  for (int mi = 0; mi < 2; ++mi) {
    #pragma unroll
    for (int ni = 0; ni < 4; ++ni) {
      const int col = n0 + wn + ni * 16 + fr;
      const float bias = pb[col];
      #pragma unroll
      for (int r = 0; r < 8; ++r) {
        const int row = m0 + wm + mi * 16 + rbase + r;
        float x = acc[mi][ni][r] + bias;
        float g = 0.5f * x * (1.0f + erff(x * 0.70710678118654752f));
        pe[(long)row * EMB + col] = f2bf(g);
      }
    }
  }
}

// ---------------------------------------------------------------------------
// Kernel 2: 2D RoPE, in-place on bf16 pe, packed 32-bit per pair.
// ---------------------------------------------------------------------------
__global__ __launch_bounds__(256) void k_rope(
    unsigned short* __restrict__ pe,
    const float* __restrict__ cx, const float* __restrict__ sx,
    const float* __restrict__ cy, const float* __restrict__ sy)
{
  const int tid = blockIdx.x * 256 + threadIdx.x;  // NTOK*512 threads
  const int t   = tid >> 9;
  const int j   = tid & 511;
  const int pid = t & 4095;
  int col, ti;
  const float *ct, *st;
  if (j < 256) {
    col = 2 * j;            ct = cx; st = sx;
    ti  = (pid & 63) * 256 + j;
  } else {
    const int j2 = j - 256;
    col = 512 + 2 * j2;     ct = cy; st = sy;
    ti  = (pid >> 6) * 256 + j2;
  }
  const float c = ct[ti], s = st[ti];
  unsigned int* p = (unsigned int*)(pe + (long)t * EMB + col);  // col even
  const unsigned int u = *p;
  const float x1 = bf_lo(u);
  const float x2 = bf_hi(u);
  *p = pack2(x1 * c - x2 * s, x1 * s + x2 * c);
}

// ---------------------------------------------------------------------------
// Kernel 3: merge GEMM  [8192 x 4096] * [4096 x 2048] + bias -> f32 out.
// Async global->LDS staging (CDNA5 ASYNCcnt path) when available.
// ---------------------------------------------------------------------------
__global__ __launch_bounds__(256) void k_merge(
    const unsigned short* __restrict__ pe,   // [NTOK][EMB] bf16
    const unsigned short* __restrict__ dwb,  // [NLLM][KMRG] bf16, K = pq*1024+e
    const float* __restrict__ db,            // [NLLM]
    float* __restrict__ out)                 // [MTOK][NLLM]
{
#if USE_ASYNC_LDS
  __shared__ __align__(16) unsigned short As[3][128][32];
  __shared__ __align__(16) unsigned short Bs[3][128][32];
#else
  __shared__ __align__(16) unsigned short As[2][128][32];
  __shared__ __align__(16) unsigned short Bs[2][128][32];
#endif

  const int tid  = threadIdx.x;
  const int lane = tid & 31;
  const int wid  = tid >> 5;
  const int m0   = blockIdx.x * 128;
  const int n0   = blockIdx.y * 128;
  const int wm   = (wid & 3) * 32;
  const int wn   = (wid >> 2) * 64;
  const int lrow = tid >> 1;
  const int lk   = (tid & 1) * 16;

  const int gm   = m0 + lrow;
  const int bimg = gm >> 10;
  const int hh   = (gm >> 5) & 31;
  const int ww   = gm & 31;
  const unsigned short* browp = dwb + (long)(n0 + lrow) * KMRG;

  v8f acc[2][4];
  #pragma unroll
  for (int mi = 0; mi < 2; ++mi)
    #pragma unroll
    for (int ni = 0; ni < 4; ++ni)
      acc[mi][ni] = (v8f){0.f,0.f,0.f,0.f,0.f,0.f,0.f,0.f};

  const int fr = lane & 15;
  const int kh = (lane >> 4) * 8;

  auto aptr = [&](int k0) -> const unsigned short* {
    const int s  = k0 + lk;                 // never crosses a 1024 block
    const int pq = s >> 10;
    const int e0 = s & 1023;
    const long srow = (long)bimg * 4096 +
                      (2 * hh + (pq >> 1)) * 64 + (2 * ww + (pq & 1));
    return pe + srow * EMB + e0;
  };

  auto compute = [&](int buf) {
    Frag af[2], bfr[4];
    #pragma unroll
    for (int mi = 0; mi < 2; ++mi) {
      const unsigned short* sp = &As[buf][wm + mi * 16 + fr][0];
      af[mi].q[0] = *(const uint4*)(sp + kh);
      af[mi].q[1] = *(const uint4*)(sp + 16 + kh);
    }
    #pragma unroll
    for (int ni = 0; ni < 4; ++ni) {
      const unsigned short* sp = &Bs[buf][wn + ni * 16 + fr][0];
      bfr[ni].q[0] = *(const uint4*)(sp + kh);
      bfr[ni].q[1] = *(const uint4*)(sp + 16 + kh);
    }
    #pragma unroll
    for (int mi = 0; mi < 2; ++mi)
      #pragma unroll
      for (int ni = 0; ni < 4; ++ni)
        acc[mi][ni] = __builtin_amdgcn_wmma_f32_16x16x32_bf16(
            false, af[mi].v, false, bfr[ni].v, (short)0, acc[mi][ni],
            false, false);
  };

  const int NC = KMRG / 32;                // 128

#if USE_ASYNC_LDS
  auto issue = [&](int kc, int buf) {      // 4 async b128 copies per thread
    const unsigned short* ga = aptr(kc * 32);
    const unsigned short* gb = browp + kc * 32 + lk;
    __builtin_amdgcn_global_load_async_to_lds_b128(
        (g_v4i)(ga),     (l_v4i)&As[buf][lrow][lk],     0, 0);
    __builtin_amdgcn_global_load_async_to_lds_b128(
        (g_v4i)(ga + 8), (l_v4i)&As[buf][lrow][lk + 8], 0, 0);
    __builtin_amdgcn_global_load_async_to_lds_b128(
        (g_v4i)(gb),     (l_v4i)&Bs[buf][lrow][lk],     0, 0);
    __builtin_amdgcn_global_load_async_to_lds_b128(
        (g_v4i)(gb + 8), (l_v4i)&Bs[buf][lrow][lk + 8], 0, 0);
  };

  issue(0, 0);
  for (int kc = 0; kc < NC - 1; ++kc) {
    issue(kc + 1, (kc + 1) % 3);
    __builtin_amdgcn_s_wait_asynccnt(4);   // in-order: chunk kc's 4 ops done
    __syncthreads();
    compute(kc % 3);
  }
  __builtin_amdgcn_s_wait_asynccnt(0);
  __syncthreads();
  compute((NC - 1) % 3);
#else
  uint4 ra[2], rb[2];
  {
    const uint4* pa = (const uint4*)aptr(0);
    const uint4* pbq = (const uint4*)(browp + lk);
    ra[0] = pa[0]; ra[1] = pa[1];
    rb[0] = pbq[0]; rb[1] = pbq[1];
  }
  for (int kc = 0; kc < NC; ++kc) {
    const int buf = kc & 1;
    *(uint4*)&As[buf][lrow][lk]     = ra[0];
    *(uint4*)&As[buf][lrow][lk + 8] = ra[1];
    *(uint4*)&Bs[buf][lrow][lk]     = rb[0];
    *(uint4*)&Bs[buf][lrow][lk + 8] = rb[1];
    __syncthreads();
    if (kc + 1 < NC) {
      const uint4* pa = (const uint4*)aptr((kc + 1) * 32);
      const uint4* pbq = (const uint4*)(browp + (kc + 1) * 32 + lk);
      ra[0] = pa[0]; ra[1] = pa[1];
      rb[0] = pbq[0]; rb[1] = pbq[1];
    }
    compute(buf);
  }
#endif

  const int rbase = (lane >> 4) * 8;
  #pragma unroll
  for (int mi = 0; mi < 2; ++mi) {
    #pragma unroll
    for (int ni = 0; ni < 4; ++ni) {
      const int col = n0 + wn + ni * 16 + fr;
      const float bias = db[col];
      #pragma unroll
      for (int r = 0; r < 8; ++r) {
        const int row = m0 + wm + mi * 16 + rbase + r;
        out[(long)row * NLLM + col] = acc[mi][ni][r] + bias;
      }
    }
  }
}

// ---------------------------------------------------------------------------
extern "C" void kernel_launch(void* const* d_in, const int* in_sizes, int n_in,
                              void* d_out, int out_size, void* d_ws, size_t ws_size,
                              hipStream_t stream) {
  (void)in_sizes; (void)n_in; (void)out_size; (void)ws_size;
  const float* pix = (const float*)d_in[0];
  // d_in[1] = grid_hw (int32), fixed 64x64 per setup_inputs
  const float* pw  = (const float*)d_in[2];
  const float* pb  = (const float*)d_in[3];
  const float* dw  = (const float*)d_in[4];
  const float* db  = (const float*)d_in[5];
  const float* cx  = (const float*)d_in[6];
  const float* sx  = (const float*)d_in[7];
  const float* cy  = (const float*)d_in[8];
  const float* sy  = (const float*)d_in[9];
  float* out = (float*)d_out;

  unsigned short* pe  = (unsigned short*)d_ws;                     // 64 MiB
  unsigned short* dwb = (unsigned short*)((char*)d_ws +
                        (size_t)NTOK * EMB * sizeof(unsigned short)); // +16 MiB

  dim3 blk(256);
  k_convert_dw<<<dim3((NLLM * 4 * 128) / 256), blk, 0, stream>>>(dw, dwb);
  k_patch_embed<<<dim3(NTOK / 128, EMB / 128), blk, 0, stream>>>(pix, pw, pb, pe);
  k_rope<<<dim3((NTOK * 512) / 256), blk, 0, stream>>>(pe, cx, sx, cy, sy);
  k_merge<<<dim3(MTOK / 128, NLLM / 128), blk, 0, stream>>>(pe, dwb, db, out);
}
